// WordLSTMEncoder_15522011807801
// MI455X (gfx1250) — matrix operand order repaired and checked by reading, not compiled, not run on hardware
//
#include <hip/hip_runtime.h>
#include <hip/hip_bf16.h>

#define BATCH 64
#define SEQT  256
#define T1    257
#define HID   1024
#define G4    4096   /* 4*H */
#define DIN   1024
#define MTOT  (BATCH * T1)   /* 16448 = 257*64 */

typedef __bf16 bf16;
typedef __attribute__((ext_vector_type(16))) __bf16 v16bf;
typedef __attribute__((ext_vector_type(8)))  float  v8f;
typedef __attribute__((ext_vector_type(4)))  unsigned int u32x4;

union FragBF { v16bf v; u32x4 q[2]; };

__device__ __forceinline__ bf16 f2bf(float f) {
  unsigned int u = __float_as_uint(f);
  u += 0x7fffu + ((u >> 16) & 1u);          // round-to-nearest-even
  unsigned short s = (unsigned short)(u >> 16);
  bf16 r;
  __builtin_memcpy(&r, &s, 2);
  return r;
}

__device__ __forceinline__ float sigm(float x) {
  return 1.0f / (1.0f + __expf(-x));
}

// ---------------------------------------------------------------------------
// Cast fp32 -> bf16 (weights)
// ---------------------------------------------------------------------------
__global__ void k_cast_bf16(const float* __restrict__ src,
                            bf16* __restrict__ dst, long long n) {
  long long i = (long long)blockIdx.x * blockDim.x + threadIdx.x;
  if (i < n) dst[i] = f2bf(src[i]);
}

// ---------------------------------------------------------------------------
// Build xr = [root ; x] as bf16, shape [B, T1, DIN]
// ---------------------------------------------------------------------------
__global__ void k_build_xr(const float* __restrict__ x,
                           const float* __restrict__ root,
                           bf16* __restrict__ xr) {
  long long idx = (long long)blockIdx.x * blockDim.x + threadIdx.x;
  long long total = (long long)BATCH * T1 * DIN;
  if (idx >= total) return;
  int d = (int)(idx % DIN);
  long long bt = idx / DIN;
  int t = (int)(bt % T1);
  int b = (int)(bt / T1);
  float v = (t == 0) ? root[d]
                     : x[((long long)b * SEQT + (t - 1)) * DIN + d];
  xr[idx] = f2bf(v);
}

// ---------------------------------------------------------------------------
// Zero recurrent state: h,c fp32 and h_bf bf16, each [2][B][H] = 131072 elems
// ---------------------------------------------------------------------------
__global__ void k_init_state(float* __restrict__ h_st, float* __restrict__ c_st,
                             bf16* __restrict__ h_bf) {
  int i = blockIdx.x * blockDim.x + threadIdx.x;
  h_st[i] = 0.0f;
  c_st[i] = 0.0f;
  h_bf[i] = f2bf(0.0f);
}

// ---------------------------------------------------------------------------
// Input projection: pre[dir][r][n] = A[r] . W[dir][n] + bias[dir][n]
// A: [Mtot, K] bf16 row-major; W: [2][G4][K] bf16 row-major; pre fp32.
// Wave computes a 64x64 tile (4 Mx16 x 4 Nx16 subtiles), K step 32.
// grid = (G4/512, Mtot/64, 2), block = 256 (8 waves, each 64 cols)
// ---------------------------------------------------------------------------
__global__ __launch_bounds__(256) void k_input_gemm(
    const bf16* __restrict__ A,
    const bf16* __restrict__ W,
    const float* __restrict__ bias,
    float* __restrict__ pre,
    int K, int Mtot)
{
  const int dir  = blockIdx.z;
  const int wave = threadIdx.x >> 5;
  const int lane = threadIdx.x & 31;
  const int half = lane >> 4;
  const int lr   = lane & 15;
  const int nbase = blockIdx.x * 512 + wave * 64;
  const int mbase = blockIdx.y * 64;
  const bf16* Bw = W + (size_t)dir * G4 * K;

  const v8f vz = {0.f,0.f,0.f,0.f,0.f,0.f,0.f,0.f};
  v8f acc[4][4];
#pragma unroll
  for (int mi = 0; mi < 4; ++mi)
#pragma unroll
    for (int ni = 0; ni < 4; ++ni) acc[mi][ni] = vz;

  for (int k0 = 0; k0 < K; k0 += 32) {
    FragBF bfr[4];
#pragma unroll
    for (int ni = 0; ni < 4; ++ni) {
      const bf16* p = Bw + (size_t)(nbase + ni * 16 + lr) * K + k0 + half * 16;
      bfr[ni].q[0] = *(const u32x4*)(p);
      bfr[ni].q[1] = *(const u32x4*)(p + 8);
    }
#pragma unroll
    for (int mi = 0; mi < 4; ++mi) {
      const bf16* p = A + (size_t)(mbase + mi * 16 + lr) * K + k0 + half * 16;
      FragBF af;
      af.q[0] = *(const u32x4*)(p);
      af.q[1] = *(const u32x4*)(p + 8);
#pragma unroll
      for (int ni = 0; ni < 4; ++ni)
        acc[mi][ni] = __builtin_amdgcn_wmma_f32_16x16x32_bf16(
            false, af.v, false, bfr[ni].v, (short)0, acc[mi][ni], false, false);
    }
  }

  const float* bp = bias + (size_t)dir * G4;
  float* pp = pre + (size_t)dir * Mtot * G4;
#pragma unroll
  for (int mi = 0; mi < 4; ++mi)
#pragma unroll
    for (int ni = 0; ni < 4; ++ni)
#pragma unroll
      for (int r = 0; r < 8; ++r) {
        int m = mbase + mi * 16 + half * 8 + r;
        int n = nbase + ni * 16 + lr;
        pp[(size_t)m * G4 + n] = acc[mi][ni][r] + bp[n];
      }
}

// ---------------------------------------------------------------------------
// One recurrent step, both directions: g[dir][b][n] = h_bf[dir] . Whh[dir][n]
//                                                   + pre[dir][b*T1+time][n]
// dir0 processes time=t, dir1 processes time=T1-1-t.
// The 128 KB h-state matrix (A) is identical for all 8 waves of a WG, so it
// is staged once into LDS with gfx1250 async global->LDS copies (ASYNCcnt),
// then read as A fragments via ds_load_b128.
// grid = (G4/512, 1, 2), block = 256 (8 waves x 64 cols, M=64 in-wave)
// ---------------------------------------------------------------------------
__global__ __launch_bounds__(256) void k_step_gemm(
    const bf16* __restrict__ h_bf,   // [2][B][H]
    const bf16* __restrict__ Whh,    // [2][G4][H]
    const float* __restrict__ pre,   // [2][Mtot][G4]
    float* __restrict__ g,           // [2][B][G4]
    int t)
{
  __shared__ bf16 As[BATCH * HID];   // 64 x 1024 bf16 = 128 KB

  const int dir  = blockIdx.z;
  const int wave = threadIdx.x >> 5;
  const int lane = threadIdx.x & 31;
  const int half = lane >> 4;
  const int lr   = lane & 15;
  const int nbase = blockIdx.x * 512 + wave * 64;
  const int time  = dir ? (T1 - 1 - t) : t;
  const bf16* A  = h_bf + (size_t)dir * BATCH * HID;
  const bf16* Bw = Whh  + (size_t)dir * G4 * HID;

  // ---- async stage A into LDS: 256 threads x 16 B x 32 iters = 128 KB ----
  {
    const unsigned lds_base = (unsigned)(size_t)(&As[0]);
    const unsigned byte0 = threadIdx.x * 16u;
#pragma unroll 4
    for (int it = 0; it < 32; ++it) {
      unsigned boff = byte0 + (unsigned)it * 4096u;
      unsigned lds_addr = lds_base + boff;
      const bf16* gp = A + (boff >> 1);
      asm volatile("global_load_async_to_lds_b128 %0, %1, off"
                   :: "v"(lds_addr), "v"(gp) : "memory");
    }
    asm volatile("s_wait_asynccnt 0x0" ::: "memory");
    __syncthreads();
  }

  const v8f vz = {0.f,0.f,0.f,0.f,0.f,0.f,0.f,0.f};
  v8f acc[4][4];
#pragma unroll
  for (int mi = 0; mi < 4; ++mi)
#pragma unroll
    for (int ni = 0; ni < 4; ++ni) acc[mi][ni] = vz;

  for (int k0 = 0; k0 < HID; k0 += 32) {
    FragBF bfr[4];
#pragma unroll
    for (int ni = 0; ni < 4; ++ni) {
      const bf16* p = Bw + (size_t)(nbase + ni * 16 + lr) * HID + k0 + half * 16;
      bfr[ni].q[0] = *(const u32x4*)(p);
      bfr[ni].q[1] = *(const u32x4*)(p + 8);
    }
#pragma unroll
    for (int mi = 0; mi < 4; ++mi) {
      const bf16* p = &As[(size_t)(mi * 16 + lr) * HID + k0 + half * 16];
      FragBF af;
      af.q[0] = *(const u32x4*)(p);          // ds_load_b128
      af.q[1] = *(const u32x4*)(p + 8);      // ds_load_b128
#pragma unroll
      for (int ni = 0; ni < 4; ++ni)
        acc[mi][ni] = __builtin_amdgcn_wmma_f32_16x16x32_bf16(
            false, af.v, false, bfr[ni].v, (short)0, acc[mi][ni], false, false);
    }
  }

  const float* pp = pre + (size_t)dir * MTOT * G4;
  float* gp = g + (size_t)dir * BATCH * G4;
#pragma unroll
  for (int mi = 0; mi < 4; ++mi)
#pragma unroll
    for (int ni = 0; ni < 4; ++ni)
#pragma unroll
      for (int r = 0; r < 8; ++r) {
        int b = mi * 16 + half * 8 + r;
        int n = nbase + ni * 16 + lr;
        gp[(size_t)b * G4 + n] =
            acc[mi][ni][r] + pp[((size_t)b * T1 + time) * G4 + n];
      }
}

// ---------------------------------------------------------------------------
// Gate pointwise + state update + output write, both directions.
// 2*B*H = 131072 threads. grid = 512, block = 256.
// ---------------------------------------------------------------------------
__global__ void k_step_point(const float* __restrict__ g,      // [2][B][G4]
                             float* __restrict__ c_st,         // [2][B][H]
                             float* __restrict__ h_st,         // [2][B][H]
                             bf16*  __restrict__ h_bf,         // [2][B][H]
                             const int* __restrict__ lengths,  // [B]
                             bf16*  __restrict__ out_bf,       // [B][T1][2H] or null
                             float* __restrict__ out_f32,      // [B][T1][2H] or null
                             int t)
{
  int idx = blockIdx.x * blockDim.x + threadIdx.x;   // < 2*64*1024
  int dir = idx >> 16;
  int rem = idx & 0xFFFF;
  int b = rem >> 10;
  int j = rem & 1023;
  int time = dir ? (T1 - 1 - t) : t;

  size_t gb = (size_t)dir * BATCH * G4 + (size_t)b * G4;
  float iv = sigm(g[gb + j]);
  float fv = sigm(g[gb + 1024 + j]);
  float gv = tanhf(g[gb + 2048 + j]);
  float ov = sigm(g[gb + 3072 + j]);

  size_t sidx = (size_t)idx;
  float c_old = c_st[sidx];
  float h_old = h_st[sidx];
  float c_new = fv * c_old + iv * gv;
  float h_new = ov * tanhf(c_new);

  bool msk = time < (lengths[b] + 1);
  float c_sel = msk ? c_new : c_old;
  float h_sel = msk ? h_new : h_old;
  c_st[sidx] = c_sel;
  h_st[sidx] = h_sel;
  h_bf[sidx] = f2bf(h_sel);

  float outv = msk ? h_new : 0.0f;
  size_t oidx = ((size_t)b * T1 + time) * 2048 + (size_t)dir * 1024 + j;
  if (out_bf)  out_bf[oidx]  = f2bf(outv);
  if (out_f32) out_f32[oidx] = outv;
}

// ---------------------------------------------------------------------------
// Host side
// ---------------------------------------------------------------------------
static inline size_t align256(size_t x) { return (x + 255) & ~(size_t)255; }

extern "C" void kernel_launch(void* const* d_in, const int* in_sizes, int n_in,
                              void* d_out, int out_size, void* d_ws, size_t ws_size,
                              hipStream_t stream) {
  (void)in_sizes; (void)n_in; (void)out_size; (void)ws_size;

  const float* x        = (const float*)d_in[0];   // [B, 256, 1024]
  const int*   lengths  = (const int*)  d_in[1];   // [B]
  const float* root     = (const float*)d_in[2];   // [1024]
  const float* Wih0     = (const float*)d_in[3];   // [2, 4096, 1024]
  const float* Whh0     = (const float*)d_in[4];   // [2, 4096, 1024]
  const float* b0       = (const float*)d_in[5];   // [2, 4096]
  const float* Wih_rest = (const float*)d_in[6];   // [2, 2, 4096, 2048]
  const float* Whh_rest = (const float*)d_in[7];   // [2, 2, 4096, 1024]
  const float* b_rest   = (const float*)d_in[8];   // [2, 2, 4096]
  float* out = (float*)d_out;                      // [B, T1, 2048]

  // ---- workspace layout ----
  char* ws = (char*)d_ws;
  size_t off = 0;

  const size_t n_wih0 = 2ull * G4 * 1024;          //  8,388,608
  const size_t n_whh0 = 2ull * G4 * 1024;          //  8,388,608
  const size_t n_wihr = 2ull * 2 * G4 * 2048;      // 33,554,432
  const size_t n_whhr = 2ull * 2 * G4 * 1024;      // 16,777,216

  bf16* wih0_bf = (bf16*)(ws + off);
  bf16* whh0_bf = wih0_bf + n_wih0;
  bf16* wihr_bf = whh0_bf + n_whh0;
  bf16* whhr_bf = wihr_bf + n_wihr;
  off = align256(off + (n_wih0 + n_whh0 + n_wihr + n_whhr) * sizeof(bf16));

  bf16* xr_bf = (bf16*)(ws + off);
  off = align256(off + (size_t)MTOT * DIN * sizeof(bf16));

  float* pre = (float*)(ws + off);                 // [2][MTOT][G4] fp32
  off = align256(off + 2ull * MTOT * G4 * sizeof(float));

  bf16* hbuf0 = (bf16*)(ws + off);                 // [B][T1][2H] bf16
  off = align256(off + (size_t)MTOT * 2048 * sizeof(bf16));
  bf16* hbuf1 = (bf16*)(ws + off);
  off = align256(off + (size_t)MTOT * 2048 * sizeof(bf16));

  float* h_st = (float*)(ws + off); off = align256(off + 2ull*BATCH*HID*sizeof(float));
  float* c_st = (float*)(ws + off); off = align256(off + 2ull*BATCH*HID*sizeof(float));
  bf16*  h_bf = (bf16*)(ws + off);  off = align256(off + 2ull*BATCH*HID*sizeof(bf16));
  float* gbuf = (float*)(ws + off); off = align256(off + 2ull*BATCH*G4*sizeof(float));

  // ---- cast weights to bf16 ----
  auto cast = [&](const float* s, bf16* d, long long n) {
    int grid = (int)((n + 255) / 256);
    hipLaunchKernelGGL(k_cast_bf16, dim3(grid), dim3(256), 0, stream, s, d, n);
  };
  cast(Wih0,     wih0_bf, (long long)n_wih0);
  cast(Whh0,     whh0_bf, (long long)n_whh0);
  cast(Wih_rest, wihr_bf, (long long)n_wihr);
  cast(Whh_rest, whhr_bf, (long long)n_whhr);

  // ---- build xr ----
  {
    long long n = (long long)MTOT * DIN;
    int grid = (int)((n + 255) / 256);
    hipLaunchKernelGGL(k_build_xr, dim3(grid), dim3(256), 0, stream, x, root, xr_bf);
  }

  // ---- layers ----
  for (int l = 0; l < 3; ++l) {
    const bf16*  in_bf = (l == 0) ? xr_bf : (l == 1) ? hbuf0 : hbuf1;
    const int    K     = (l == 0) ? 1024 : 2048;
    const bf16*  Wih_l = (l == 0) ? wih0_bf : wihr_bf + (size_t)(l - 1) * 2 * G4 * 2048;
    const bf16*  Whh_l = (l == 0) ? whh0_bf : whhr_bf + (size_t)(l - 1) * 2 * G4 * 1024;
    const float* b_l   = (l == 0) ? b0      : b_rest  + (size_t)(l - 1) * 2 * G4;
    bf16*  out_bf  = (l == 0) ? hbuf0 : (l == 1) ? hbuf1 : (bf16*)nullptr;
    float* out_f32 = (l == 2) ? out : (float*)nullptr;

    // input projection over all timesteps, both directions
    hipLaunchKernelGGL(k_input_gemm, dim3(G4 / 512, MTOT / 64, 2), dim3(256), 0,
                       stream, in_bf, Wih_l, b_l, pre, K, MTOT);

    // zero recurrent state
    hipLaunchKernelGGL(k_init_state, dim3(512), dim3(256), 0, stream,
                       h_st, c_st, h_bf);

    // sequential recurrence; fwd and bwd fused per step
    for (int t = 0; t < T1; ++t) {
      hipLaunchKernelGGL(k_step_gemm, dim3(G4 / 512, 1, 2), dim3(256), 0,
                         stream, h_bf, Whh_l, pre, gbuf, t);
      hipLaunchKernelGGL(k_step_point, dim3(512), dim3(256), 0,
                         stream, gbuf, c_st, h_st, h_bf, lengths,
                         out_bf, out_f32, t);
    }
  }
}
